// AttentionEncoder_39865886441506
// MI455X (gfx1250) — compile-verified, matrix operand order checked by reading, not run on hardware
//
#include <hip/hip_runtime.h>
#include <hip/hip_bf16.h>

// Problem constants (match reference)
constexpr int BN = 16384;  // batch
constexpr int TN = 48;     // lags
constexpr int DN = 32;     // features
constexpr int HN = 64;     // hidden

typedef _Float16 half16 __attribute__((ext_vector_type(16)));
typedef float    float8 __attribute__((ext_vector_type(8)));

// Fast transcendentals: single v_exp_f32 + single v_rcp_f32 (no IEEE div chain)
__device__ __forceinline__ float sigf(float x) {
    return __builtin_amdgcn_rcpf(1.0f + __expf(-x));
}
__device__ __forceinline__ float tanhfast(float x) {
    // tanh(x) = 1 - 2/(exp(2x)+1); saturates correctly at +/-1
    return 1.0f - 2.0f * __builtin_amdgcn_rcpf(__expf(2.0f * x) + 1.0f);
}

// ---------------------------------------------------------------------------
// Kernel 0: time-invariant input attention.
//   The h/c parts of the attention logit are constant over the feature axis,
//   so they cancel in the softmax:  a[b,d] = softmax_d( sum_t w_x[t]*x[b,t,d] )
//   and a is identical for every timestep.
//   attention[b,t,d] = a[b,d] for all t;  ws_a[b,d] = a[b,d]
// One wave per batch row; lane == feature index (D == wave32 width).
// ---------------------------------------------------------------------------
__global__ __launch_bounds__(256) void
attn_kernel(const float* __restrict__ x, const float* __restrict__ W_att,
            float* __restrict__ att_out, float* __restrict__ ws_a) {
    const int wave = (blockIdx.x * blockDim.x + threadIdx.x) >> 5;
    const int lane = threadIdx.x & 31;
    if (wave >= BN) return;

    const float* xb = x + (size_t)wave * TN * DN;
    float s = 0.0f;
#pragma unroll
    for (int t = 0; t < TN; ++t) {
        s += W_att[2 * HN + t] * xb[t * DN + lane];   // uniform weight -> s_load
    }
    // wave-wide softmax over 32 lanes
    float m = s;
#pragma unroll
    for (int off = 16; off >= 1; off >>= 1)
        m = fmaxf(m, __shfl_xor(m, off, 32));
    float e = __expf(s - m);
    float sum = e;
#pragma unroll
    for (int off = 16; off >= 1; off >>= 1)
        sum += __shfl_xor(sum, off, 32);
    const float a = e * __builtin_amdgcn_rcpf(sum);

    ws_a[(size_t)wave * DN + lane] = a;
    float* ao = att_out + (size_t)wave * TN * DN;
#pragma unroll
    for (int t = 0; t < TN; ++t)
        ao[t * DN + lane] = a;   // 128B coalesced per store
}

// ---------------------------------------------------------------------------
// Kernel 1: sequential LSTM scan with WMMA.
// Block = 16 batch rows, 4 waves. Wave w owns h-columns [16w,16w+16) of all
// four gates. 12x v_wmma_f32_16x16x32_f16 per wave per step; weights live in
// VGPRs for the whole scan; h is exchanged through LDS each step.
//
// 16-bit A operand layout (16x32, row M = lane%16):
//   half i in [0,8):  K = (lane/16)*8 + i
//   half i in [8,16): K = (lane/16)*8 + 16 + (i-8)
// B operand (32x16, col N = lane%16) uses the same K pattern.
// C/D: element j -> (M = j + 8*(lane/16), N = lane%16).
// ---------------------------------------------------------------------------
__global__ __launch_bounds__(128, 1) void
lstm_kernel(const float* __restrict__ x, const float* __restrict__ h0,
            const float* __restrict__ c0, const float* __restrict__ W_ih,
            const float* __restrict__ W_hh, const float* __restrict__ b_ih,
            const float* __restrict__ b_hh, const float* __restrict__ ws_a,
            float* __restrict__ enc_out) {
    __shared__ _Float16 h_lds[16 * HN];   // 2KB: h tile in f16, row-major [16][64]

    const int lane  = threadIdx.x & 31;
    const int w     = threadIdx.x >> 5;   // 0..3  (gate-column tile)
    const int row16 = lane & 15;
    const int hs    = lane >> 4;          // A/B half-select
    const int n0    = w * 16;
    const int b0    = blockIdx.x * 16;
    const int hcol  = n0 + row16;         // this lane's h/gate column
    const int kb    = hs * 8;             // K base for this lane

    // ---- Load all weight B-operands into registers (f32 -> f16) ----
    half16 Bih[4], Bhh0[4], Bhh1[4];
    float  biasv[4];
#pragma unroll
    for (int g = 0; g < 4; ++g) {
        const int ng = g * HN + hcol;                  // gate row in [0,256)
        const float* wr = W_ih + (size_t)ng * DN;      // row of W_ih [256,32]
        const float* hr = W_hh + (size_t)ng * HN;      // row of W_hh [256,64]
#pragma unroll
        for (int i = 0; i < 8; ++i) {
            Bih[g][i]      = (_Float16)wr[kb + i];
            Bih[g][i + 8]  = (_Float16)wr[kb + 16 + i];
            Bhh0[g][i]     = (_Float16)hr[kb + i];
            Bhh0[g][i + 8] = (_Float16)hr[kb + 16 + i];
            Bhh1[g][i]     = (_Float16)hr[32 + kb + i];
            Bhh1[g][i + 8] = (_Float16)hr[32 + kb + 16 + i];
        }
        biasv[g] = b_ih[ng] + b_hh[ng];
    }

    // ---- Per-lane attention weights for the wi A-operand ----
    float ak[16];
    {
        const float* ar = ws_a + (size_t)(b0 + row16) * DN;
#pragma unroll
        for (int i = 0; i < 8; ++i) {
            ak[i]     = ar[kb + i];
            ak[i + 8] = ar[kb + 16 + i];
        }
    }

    // ---- Persistent cell state (C/D layout) ----
    float c_reg[8];
#pragma unroll
    for (int j = 0; j < 8; ++j)
        c_reg[j] = c0[(size_t)(b0 + j + 8 * hs) * HN + hcol];

    // ---- Initial h A-operands from h0 ----
    half16 hA0, hA1;
    {
        const float* h0r = h0 + (size_t)(b0 + row16) * HN;
#pragma unroll
        for (int i = 0; i < 8; ++i) {
            hA0[i]     = (_Float16)h0r[kb + i];
            hA0[i + 8] = (_Float16)h0r[kb + 16 + i];
            hA1[i]     = (_Float16)h0r[32 + kb + i];
            hA1[i + 8] = (_Float16)h0r[32 + kb + 16 + i];
        }
    }

    const float* xr = x + (size_t)(b0 + row16) * TN * DN;

    for (int t = 0; t < TN; ++t) {
        // wi = a .* x_t, packed into A-operand layout
        const float* xt = xr + t * DN;
        if (t + 1 < TN) __builtin_prefetch(xt + DN, 0, 0);
        half16 wiA;
#pragma unroll
        for (int i = 0; i < 8; ++i) {
            wiA[i]     = (_Float16)(ak[i]     * xt[kb + i]);
            wiA[i + 8] = (_Float16)(ak[i + 8] * xt[kb + 16 + i]);
        }

        // gates = bias + wi @ W_ih^T + h @ W_hh^T  (12 WMMAs)
        float8 acc[4];
#pragma unroll
        for (int g = 0; g < 4; ++g) {
            float8 a8;
#pragma unroll
            for (int j = 0; j < 8; ++j) a8[j] = biasv[g];
            a8 = __builtin_amdgcn_wmma_f32_16x16x32_f16(false, wiA, false, Bih[g],
                                                        (short)0, a8, false, false);
            a8 = __builtin_amdgcn_wmma_f32_16x16x32_f16(false, hA0, false, Bhh0[g],
                                                        (short)0, a8, false, false);
            a8 = __builtin_amdgcn_wmma_f32_16x16x32_f16(false, hA1, false, Bhh1[g],
                                                        (short)0, a8, false, false);
            acc[g] = a8;
        }

        // LSTM elementwise update; write h to output + LDS (f16)
#pragma unroll
        for (int j = 0; j < 8; ++j) {
            const float ig = sigf(acc[0][j]);
            const float fg = sigf(acc[1][j]);
            const float gg = tanhfast(acc[2][j]);
            const float og = sigf(acc[3][j]);
            const float c  = fg * c_reg[j] + ig * gg;
            c_reg[j] = c;
            const float h = og * tanhfast(c);
            enc_out[((size_t)(b0 + j + 8 * hs) * TN + t) * HN + hcol] = h;
            h_lds[(j + 8 * hs) * HN + hcol] = (_Float16)h;
        }
        __syncthreads();

        // Re-load h A-operands for next step (contiguous 16B chunks -> ds_load_b128)
        const _Float16* hr = &h_lds[row16 * HN];
#pragma unroll
        for (int i = 0; i < 8; ++i) {
            hA0[i]     = hr[kb + i];
            hA0[i + 8] = hr[kb + 16 + i];
            hA1[i]     = hr[32 + kb + i];
            hA1[i + 8] = hr[32 + kb + 16 + i];
        }
        __syncthreads();
    }
}

extern "C" void kernel_launch(void* const* d_in, const int* in_sizes, int n_in,
                              void* d_out, int out_size, void* d_ws, size_t ws_size,
                              hipStream_t stream) {
    const float* x     = (const float*)d_in[0];
    const float* h0    = (const float*)d_in[1];
    const float* c0    = (const float*)d_in[2];
    const float* W_att = (const float*)d_in[3];
    // d_in[4] = b_att: additive constant, cancels in softmax -> unused
    const float* W_ih  = (const float*)d_in[5];
    const float* W_hh  = (const float*)d_in[6];
    const float* b_ih  = (const float*)d_in[7];
    const float* b_hh  = (const float*)d_in[8];

    float* out     = (float*)d_out;
    float* att_out = out;                              // [B,T,D]
    float* enc_out = out + (size_t)BN * TN * DN;       // [B,T,H]
    float* ws_a    = (float*)d_ws;                     // [B,D] = 2MB

    attn_kernel<<<BN / 8, 256, 0, stream>>>(x, W_att, att_out, ws_a);
    lstm_kernel<<<BN / 16, 128, 0, stream>>>(x, h0, c0, W_ih, W_hh, b_ih, b_hh,
                                             ws_a, enc_out);
}
